// SetValuedAttention_69741678952866
// MI455X (gfx1250) — compile-verified, matrix operand order checked by reading, not run on hardware
//
#include <hip/hip_runtime.h>
#include <hip/hip_bf16.h>

// ---------------------------------------------------------------------------
// Problem constants (from reference)
// ---------------------------------------------------------------------------
#define BATCH      2
#define SEQ        2048
#define DMODEL     1024
#define NHEADS     16
#define DK         64
#define BS         (BATCH * SEQ)        // 4096 rows
#define BH         (BATCH * NHEADS)     // 32 (b,h) pairs
#define PHASE_BASE 7
#define PSTRIDE    (SEQ + 4)            // LDS row stride (floats); 2052%64=4 -> 16 rows hit distinct banks

typedef __attribute__((ext_vector_type(16))) _Float16 v16h;
typedef __attribute__((ext_vector_type(8)))  _Float16 v8h;
typedef __attribute__((ext_vector_type(8)))  float    v8f;
typedef __attribute__((ext_vector_type(4)))  float    v4f;

// ---------------------------------------------------------------------------
// WMMA helper: D = A(16x32 f16) * B(32x16 f16) + C (f32)
// ---------------------------------------------------------------------------
__device__ __forceinline__ v8f wmma_f32_16x16x32_f16(v16h a, v16h b, v8f c) {
  return __builtin_amdgcn_wmma_f32_16x16x32_f16(
      /*neg_a=*/false, a, /*neg_b=*/false, b,
      /*c_mod=*/(short)0, c, /*reuse_a=*/false, /*reuse_b=*/false);
}

// 16-bit A/B fragment load per documented CDNA5 layout:
// lane holds 16 halves: k = base..base+7 (contig) and base+16..base+23 (contig),
// base = (lane<16 ? 0 : 8). `p` must already point at row_start + k0 + base.
__device__ __forceinline__ v16h load_frag_f16(const _Float16* p) {
  v8h lo = *(const v8h*)(p);
  v8h hi = *(const v8h*)(p + 16);
  v16h r;
#pragma unroll
  for (int i = 0; i < 8; ++i) { r[i] = lo[i]; r[i + 8] = hi[i]; }
  return r;
}

// Same fragment but source is fp32 (works for global or LDS pointers) -> f16.
__device__ __forceinline__ v16h load_frag_f32_cvt(const float* p) {
  v4f a0 = *(const v4f*)(p);
  v4f a1 = *(const v4f*)(p + 4);
  v4f b0 = *(const v4f*)(p + 16);
  v4f b1 = *(const v4f*)(p + 20);
  v16h r;
#pragma unroll
  for (int i = 0; i < 4; ++i) {
    r[i]      = (_Float16)a0[i];
    r[i + 4]  = (_Float16)a1[i];
    r[i + 8]  = (_Float16)b0[i];
    r[i + 12] = (_Float16)b1[i];
  }
  return r;
}

// Async DMA: copy 16 bytes from LDS to global, tracked by ASYNCcnt.
// LDS offset = low 32 bits of the generic pointer (flat->LDS truncates addr[31:0]).
__device__ __forceinline__ void async_store_b128(float* gptr, const float* lds_ptr) {
  unsigned long long ga = (unsigned long long)gptr;
  unsigned lo = (unsigned)(unsigned long long)lds_ptr;
  asm volatile("global_store_async_from_lds_b128 %0, %1, off"
               :: "v"(ga), "v"(lo) : "memory");
}
__device__ __forceinline__ void wait_asynccnt0() {
  asm volatile("s_wait_asynccnt 0x0" ::: "memory");
}

__device__ __forceinline__ float phase_gain(int c, const float* g) {
  const float twopi_over_base = 6.28318530717958647692f / (float)PHASE_BASE;
  float ph = __cosf(twopi_over_base * (float)(c % PHASE_BASE));
  return g[c] * (1.0f + 0.1f * ph);
}

// ---------------------------------------------------------------------------
// K0: LDS-tiled transpose+convert: W[k][n] (f32) -> Wt[n][k] (f16).
// grid (D/32, D/32), block (32, 8)
// ---------------------------------------------------------------------------
__global__ void transpose_w_kernel(const float* __restrict__ W,
                                   _Float16* __restrict__ Wt) {
  __shared__ float tile[32][33];
  int k0 = blockIdx.x * 32;
  int n0 = blockIdx.y * 32;
  int tx = threadIdx.x;            // 0..31
  int ty = threadIdx.y;            // 0..7
#pragma unroll
  for (int j = 0; j < 4; ++j) {
    int k = k0 + ty + j * 8;
    tile[ty + j * 8][tx] = W[(size_t)k * DMODEL + n0 + tx];   // coalesced
  }
  __syncthreads();
#pragma unroll
  for (int j = 0; j < 4; ++j) {
    int n = n0 + ty + j * 8;
    Wt[(size_t)n * DMODEL + k0 + tx] = (_Float16)tile[tx][ty + j * 8]; // coalesced
  }
}

// ---------------------------------------------------------------------------
// K0b: LDS-tiled f16 transpose: V row-major [b][s][c] -> Vt [b][h][kk][s].
// grid (SEQ/32, DMODEL/32, BATCH), block (32, 8)
// ---------------------------------------------------------------------------
__global__ void vt_transpose_kernel(const _Float16* __restrict__ Vrow,
                                    _Float16* __restrict__ Vt) {
  __shared__ _Float16 tile[32][33];
  int b  = blockIdx.z;
  int s0 = blockIdx.x * 32;
  int c0 = blockIdx.y * 32;
  int tx = threadIdx.x;
  int ty = threadIdx.y;
#pragma unroll
  for (int j = 0; j < 4; ++j) {
    int s = s0 + ty + j * 8;
    tile[ty + j * 8][tx] = Vrow[((size_t)(b * SEQ + s)) * DMODEL + c0 + tx];
  }
  __syncthreads();
#pragma unroll
  for (int j = 0; j < 4; ++j) {
    int c = c0 + ty + j * 8;
    int h = c >> 6, kk = c & 63;
    Vt[((size_t)(b * NHEADS + h) * DK + kk) * SEQ + s0 + tx] = tile[tx][ty + j * 8];
  }
}

// ---------------------------------------------------------------------------
// K1: modular phase norm of input row -> f16
// ---------------------------------------------------------------------------
__global__ void norm_in_kernel(const float* __restrict__ x,
                               const float* __restrict__ g,
                               _Float16* __restrict__ xn) {
  __shared__ float sred[256];
  int row = blockIdx.x;
  int tid = threadIdx.x;
  const float* xr = x + (size_t)row * DMODEL;

  float v[4];
  float ss = 0.0f;
#pragma unroll
  for (int j = 0; j < 4; ++j) {
    v[j] = xr[tid + j * 256];
    ss += v[j] * v[j];
  }
  sred[tid] = ss;
  __syncthreads();
  for (int s = 128; s > 0; s >>= 1) {
    if (tid < s) sred[tid] += sred[tid + s];
    __syncthreads();
  }
  float rms = __frsqrt_rn(sred[0] * (1.0f / DMODEL) + 1e-6f);
  _Float16* xo = xn + (size_t)row * DMODEL;
#pragma unroll
  for (int j = 0; j < 4; ++j) {
    int c = tid + j * 256;
    xo[c] = (_Float16)(v[j] * rms * phase_gain(c, g));
  }
}

// ---------------------------------------------------------------------------
// K2: generic WMMA GEMM: C[M x N] = A[M x K] * Bt[N x K]^T
// block = 128 threads (4 waves); wave w handles n-tile blockIdx.y*64 + w*16
// ---------------------------------------------------------------------------
__global__ void wmma_gemm_f16_kernel(const _Float16* __restrict__ A, int lda,
                                     const _Float16* __restrict__ Bt, int ldb,
                                     float* __restrict__ C, int ldc, int Kdim) {
  int lane = threadIdx.x & 31;
  int wave = threadIdx.x >> 5;
  int r  = lane & 15;
  int hi = lane >> 4;
  int m0 = blockIdx.x * 16;
  int n0 = blockIdx.y * 64 + wave * 16;

  const _Float16* aRow = A  + (size_t)(m0 + r) * lda + hi * 8;
  const _Float16* bRow = Bt + (size_t)(n0 + r) * ldb + hi * 8;

  v8f acc = {};
  for (int k0 = 0; k0 < Kdim; k0 += 32) {
    v16h a = load_frag_f16(aRow + k0);
    v16h b = load_frag_f16(bRow + k0);
    acc = wmma_f32_16x16x32_f16(a, b, acc);
  }
#pragma unroll
  for (int i = 0; i < 8; ++i)
    C[(size_t)(m0 + hi * 8 + i) * ldc + n0 + r] = acc[i];
}

// ---------------------------------------------------------------------------
// K3: projection epilogue: Y += bias, Givens-rotate planes 0..3, then
//   mode 0: phase-norm with g, write f16 row-major       (Q, K)
//   mode 1: no norm, write f16 row-major                 (V; transposed later)
//   mode 2: phase-norm with g, write fp32                (final output)
// ---------------------------------------------------------------------------
__global__ void epilogue_kernel(const float* __restrict__ Y,
                                const float* __restrict__ bias,
                                const float* __restrict__ theta,
                                const float* __restrict__ g,
                                _Float16* __restrict__ outH,
                                float* __restrict__ outF,
                                int mode) {
  __shared__ float sh[DMODEL];
  __shared__ float sred[256];
  int row = blockIdx.x;
  int tid = threadIdx.x;
  const float* yr = Y + (size_t)row * DMODEL;

#pragma unroll
  for (int j = 0; j < 4; ++j) {
    int c = tid + j * 256;
    sh[c] = yr[c] + bias[c];
  }
  __syncthreads();
  if (tid < 4) {                    // 4 Givens planes on dims (2p, 2p+1)
    float th = theta[tid];
    float cs = __cosf(th), sn = __sinf(th);
    float a = sh[2 * tid], b = sh[2 * tid + 1];
    sh[2 * tid]     = cs * a - sn * b;
    sh[2 * tid + 1] = sn * a + cs * b;
  }
  __syncthreads();

  float rms = 1.0f;
  if (mode != 1) {
    float ss = 0.0f;
#pragma unroll
    for (int j = 0; j < 4; ++j) {
      float v = sh[tid + j * 256];
      ss += v * v;
    }
    sred[tid] = ss;
    __syncthreads();
    for (int s = 128; s > 0; s >>= 1) {
      if (tid < s) sred[tid] += sred[tid + s];
      __syncthreads();
    }
    rms = __frsqrt_rn(sred[0] * (1.0f / DMODEL) + 1e-6f);
  }

  if (mode == 0) {
    _Float16* o = outH + (size_t)row * DMODEL;
#pragma unroll
    for (int j = 0; j < 4; ++j) {
      int c = tid + j * 256;
      o[c] = (_Float16)(sh[c] * rms * phase_gain(c, g));
    }
  } else if (mode == 1) {
    _Float16* o = outH + (size_t)row * DMODEL;
#pragma unroll
    for (int j = 0; j < 4; ++j) {
      int c = tid + j * 256;
      o[c] = (_Float16)sh[c];
    }
  } else {
    float* o = outF + (size_t)row * DMODEL;
#pragma unroll
    for (int j = 0; j < 4; ++j) {
      int c = tid + j * 256;
      o[c] = sh[c] * rms * phase_gain(c, g);
    }
  }
}

// ---------------------------------------------------------------------------
// K4 (fused): scores + softmax + P@V for one 16-query tile of one (b,h).
// Scores live entirely in LDS (16 x 2048 fp32 = 128 KB of the 320 KB WGP LDS).
// Normalized P is DMA'd LDS->HBM with GLOBAL_STORE_ASYNC_FROM_LDS_B128, which
// overlaps the P@V WMMA loop (both only read LDS); ASYNCcnt waited at the end.
// grid: (SEQ/16, BH), block = 128 threads (4 waves).
// ---------------------------------------------------------------------------
__global__ void fused_attn_kernel(const _Float16* __restrict__ Qh,
                                  const _Float16* __restrict__ Kh,
                                  const _Float16* __restrict__ Vt,
                                  float* __restrict__ attn,
                                  _Float16* __restrict__ ctx_h) {
  __shared__ float Pl[16 * PSTRIDE];   // raw scores -> exp -> normalized probs
  __shared__ float sred[128];
  __shared__ float rowMax[16];
  __shared__ float rowSum[16];

  const int tid  = threadIdx.x;
  const int lane = tid & 31;
  const int wave = tid >> 5;
  const int r    = lane & 15;
  const int hi   = lane >> 4;
  const int m0   = blockIdx.x * 16;
  const int bh   = blockIdx.y;
  const int b    = bh / NHEADS;
  const int h    = bh % NHEADS;
  const int kend   = m0 + 16;                 // keys [0, kend) are reachable
  const int kendUp = (kend + 31) & ~31;       // WMMA k granularity

  // ---- Phase 1: scores -> LDS (Q fragment is tile-invariant: hoisted) ----
  const _Float16* aRow =
      Qh + ((size_t)(b * SEQ + m0 + r)) * DMODEL + h * DK + hi * 8;
  v16h aF0 = load_frag_f16(aRow);
  v16h aF1 = load_frag_f16(aRow + 32);

  for (int t = wave; t * 16 < kend; t += 4) {
    int n0 = t * 16;
    const _Float16* bRow =
        Kh + ((size_t)(b * SEQ + n0 + r)) * DMODEL + h * DK + hi * 8;
    __builtin_prefetch(bRow + 64, 0, 1);      // global_prefetch_b8
    v8f acc = {};
    acc = wmma_f32_16x16x32_f16(aF0, load_frag_f16(bRow), acc);
    acc = wmma_f32_16x16x32_f16(aF1, load_frag_f16(bRow + 32), acc);
#pragma unroll
    for (int i = 0; i < 8; ++i) {
      int lr = i + hi * 8;                    // local row 0..15
      int m  = m0 + lr;
      int n  = n0 + r;
      float v = acc[i] * 0.125f;              // 1/sqrt(dk)
      if (n > m) v = -1e9f;
      Pl[lr * PSTRIDE + n] = v;
    }
  }
  // pad [kend, kendUp) with -1e9 so exp() makes it exactly 0
  if (kendUp > kend) {
    for (int e = tid; e < 256; e += 128) {
      int rr = e >> 4, cc = kend + (e & 15);
      Pl[rr * PSTRIDE + cc] = -1e9f;
    }
  }
  __syncthreads();

  // ---- Phase 2: softmax (8 threads per row) ----
  const int srow = tid >> 3;                  // 0..15
  const int slot = tid & 7;                   // 0..7
  {
    float mx = -3.4e38f;
    for (int c = slot; c < kendUp; c += 8)
      mx = fmaxf(mx, Pl[srow * PSTRIDE + c]);
    sred[tid] = mx;
    __syncthreads();
    if (slot < 4) sred[tid] = fmaxf(sred[tid], sred[tid + 4]);
    __syncthreads();
    if (slot < 2) sred[tid] = fmaxf(sred[tid], sred[tid + 2]);
    __syncthreads();
    if (slot == 0) rowMax[srow] = fmaxf(sred[tid], sred[tid + 1]);
    __syncthreads();
  }
  {
    float mx = rowMax[srow];
    float sum = 0.0f;
    for (int c = slot; c < kendUp; c += 8) {
      float e = __expf(Pl[srow * PSTRIDE + c] - mx);
      Pl[srow * PSTRIDE + c] = e;
      sum += e;
    }
    sred[tid] = sum;
    __syncthreads();
    if (slot < 4) sred[tid] += sred[tid + 4];
    __syncthreads();
    if (slot < 2) sred[tid] += sred[tid + 2];
    __syncthreads();
    if (slot == 0) rowSum[srow] = sred[tid] + sred[tid + 1];
    __syncthreads();
  }
  {
    // normalize in place so LDS holds final attn values (DMA'd verbatim,
    // and P@V no longer needs a 1/rowSum epilogue)
    float inv = 1.0f / rowSum[srow];
    for (int c = slot; c < kendUp; c += 8)
      Pl[srow * PSTRIDE + c] *= inv;
  }
  __syncthreads();

  // ---- Phase 2c: launch async LDS->HBM DMA of P; zero the masked triangle ----
  {
    float* base = attn + ((size_t)bh * SEQ + m0) * SEQ;
#pragma unroll 1
    for (int rr = 0; rr < 16; ++rr) {
      float* orow = base + (size_t)rr * SEQ;
      const float* lrow = &Pl[rr * PSTRIDE];
      for (int c4 = tid * 4; c4 < kendUp; c4 += 128 * 4)
        async_store_b128(orow + c4, lrow + c4);   // overlaps with Phase 3
      v4f z = {};
      for (int c4 = kendUp + tid * 4; c4 < SEQ; c4 += 128 * 4)
        *(v4f*)(orow + c4) = z;
    }
  }

  // ---- Phase 3: context = P @ V straight from LDS ----
  {
    int n0 = wave * 16;                       // dk split across 4 waves
    const _Float16* vRow = Vt + ((size_t)bh * DK + n0 + r) * SEQ + hi * 8;
    const float*    pRow = &Pl[r * PSTRIDE + hi * 8];
    __builtin_prefetch(vRow, 0, 1);
    v8f acc = {};
    for (int k0 = 0; k0 < kendUp; k0 += 32) {
      v16h a  = load_frag_f32_cvt(pRow + k0); // ds_load_b128 x4 + cvt
      v16h bb = load_frag_f16(vRow + k0);
      acc = wmma_f32_16x16x32_f16(a, bb, acc);
    }
#pragma unroll
    for (int i = 0; i < 8; ++i) {
      int lr = i + hi * 8;
      ctx_h[((size_t)(b * SEQ + m0 + lr)) * DMODEL + h * DK + n0 + r] =
          (_Float16)acc[i];
    }
  }

  // ensure async DMA finished reading LDS before workgroup teardown
  wait_asynccnt0();
}

// ---------------------------------------------------------------------------
// Host-side orchestration
// ---------------------------------------------------------------------------
extern "C" void kernel_launch(void* const* d_in, const int* in_sizes, int n_in,
                              void* d_out, int out_size, void* d_ws, size_t ws_size,
                              hipStream_t stream) {
  const float* x    = (const float*)d_in[0];
  // d_in[1] = mask (causal, implemented analytically)
  const float* Wq   = (const float*)d_in[2];
  const float* bq   = (const float*)d_in[3];
  const float* thq  = (const float*)d_in[4];
  const float* Wk   = (const float*)d_in[5];
  const float* bk   = (const float*)d_in[6];
  const float* thk  = (const float*)d_in[7];
  const float* Wv   = (const float*)d_in[8];
  const float* bv   = (const float*)d_in[9];
  const float* thv  = (const float*)d_in[10];
  const float* Wo   = (const float*)d_in[11];
  const float* bo   = (const float*)d_in[12];
  const float* tho  = (const float*)d_in[13];
  const float* g_in = (const float*)d_in[14];
  const float* g_q  = (const float*)d_in[15];
  const float* g_k  = (const float*)d_in[16];
  const float* g_out= (const float*)d_in[17];

  float* out_f = (float*)d_out;                            // [BS, D] fp32
  float* attn  = (float*)d_out + (size_t)BS * DMODEL;      // [BH, S, S] fp32

  char* ws = (char*)d_ws;
  const size_t WSZ  = (size_t)DMODEL * DMODEL * sizeof(_Float16); // 2 MB
  _Float16* Wt_q = (_Float16*)(ws + 0 * WSZ);
  _Float16* Wt_k = (_Float16*)(ws + 1 * WSZ);
  _Float16* Wt_v = (_Float16*)(ws + 2 * WSZ);
  _Float16* Wt_o = (_Float16*)(ws + 3 * WSZ);
  size_t off = 4 * WSZ;
  _Float16* xn_h = (_Float16*)(ws + off); off += (size_t)BS * DMODEL * 2; // 8 MB
  float*    Yf   = (float*)   (ws + off); off += (size_t)BS * DMODEL * 4; // 16 MB
  _Float16* Q_h  = (_Float16*)(ws + off); off += (size_t)BS * DMODEL * 2; // 8 MB
  _Float16* K_h  = (_Float16*)(ws + off); off += (size_t)BS * DMODEL * 2; // 8 MB
  _Float16* V_h  = (_Float16*)(ws + off); off += (size_t)BS * DMODEL * 2; // 8 MB (row-major)
  _Float16* Vt_h = (_Float16*)(ws + off); off += (size_t)BS * DMODEL * 2; // 8 MB (per-head T)
  _Float16* ctx_h= (_Float16*)(ws + off); off += (size_t)BS * DMODEL * 2; // 8 MB

  const int TPB = 256;

  // 0) weight transpose+convert (LDS-tiled, coalesced both sides)
  dim3 tw_grid(DMODEL / 32, DMODEL / 32);                  // (32, 32)
  dim3 tw_block(32, 8);
  transpose_w_kernel<<<tw_grid, tw_block, 0, stream>>>(Wq, Wt_q);
  transpose_w_kernel<<<tw_grid, tw_block, 0, stream>>>(Wk, Wt_k);
  transpose_w_kernel<<<tw_grid, tw_block, 0, stream>>>(Wv, Wt_v);
  transpose_w_kernel<<<tw_grid, tw_block, 0, stream>>>(Wo, Wt_o);

  // 1) input phase norm
  norm_in_kernel<<<BS, TPB, 0, stream>>>(x, g_in, xn_h);

  dim3 gemm_grid(BS / 16, DMODEL / 64);                    // (256, 16)
  dim3 gemm_block(128);

  // 2) Q projection + epilogue (rotate + phase norm -> f16)
  wmma_gemm_f16_kernel<<<gemm_grid, gemm_block, 0, stream>>>(
      xn_h, DMODEL, Wt_q, DMODEL, Yf, DMODEL, DMODEL);
  epilogue_kernel<<<BS, TPB, 0, stream>>>(Yf, bq, thq, g_q, Q_h, nullptr, 0);

  // 3) K projection + epilogue
  wmma_gemm_f16_kernel<<<gemm_grid, gemm_block, 0, stream>>>(
      xn_h, DMODEL, Wt_k, DMODEL, Yf, DMODEL, DMODEL);
  epilogue_kernel<<<BS, TPB, 0, stream>>>(Yf, bk, thk, g_k, K_h, nullptr, 0);

  // 4) V projection + epilogue (rotate, row-major f16), then tiled transpose
  wmma_gemm_f16_kernel<<<gemm_grid, gemm_block, 0, stream>>>(
      xn_h, DMODEL, Wt_v, DMODEL, Yf, DMODEL, DMODEL);
  epilogue_kernel<<<BS, TPB, 0, stream>>>(Yf, bv, thv, nullptr, V_h, nullptr, 1);
  dim3 vt_grid(SEQ / 32, DMODEL / 32, BATCH);              // (64, 32, 2)
  vt_transpose_kernel<<<vt_grid, tw_block, 0, stream>>>(V_h, Vt_h);

  // 5) fused scores + softmax + async-DMA attn write + P@V
  dim3 fa_grid(SEQ / 16, BH);                              // (128, 32)
  fused_attn_kernel<<<fa_grid, dim3(128), 0, stream>>>(Q_h, K_h, Vt_h, attn, ctx_h);

  // 6) output projection + epilogue (rotate + phase norm -> fp32 d_out)
  wmma_gemm_f16_kernel<<<gemm_grid, gemm_block, 0, stream>>>(
      ctx_h, DMODEL, Wt_o, DMODEL, Yf, DMODEL, DMODEL);
  epilogue_kernel<<<BS, TPB, 0, stream>>>(Yf, bo, tho, g_out, nullptr, out_f, 2);
}
